// RDesign_Model_81527069213137
// MI455X (gfx1250) — matrix operand order repaired and checked by reading, not compiled, not run on hardware
//
#include <hip/hip_runtime.h>
#include <hip/hip_bf16.h>

#define HD  128
#define HD3 384
#define HD4 512
#define LAY 6

typedef __attribute__((ext_vector_type(16))) _Float16 v16h;
typedef __attribute__((ext_vector_type(8)))  _Float16 v8h;
typedef __attribute__((ext_vector_type(4)))  _Float16 v4h;
typedef __attribute__((ext_vector_type(8)))  float    v8f;

__device__ __forceinline__ v8f vzero() {
  v8f z = {0.f, 0.f, 0.f, 0.f, 0.f, 0.f, 0.f, 0.f};
  return z;
}

__device__ __forceinline__ v8f wmma16(v16h a, v16h b, v8f c) {
  // D = A(16x32 f16) * B(32x16 f16) + C(16x16 f32)
  return __builtin_amdgcn_wmma_f32_16x16x32_f16(
      /*neg_a=*/false, a, /*neg_b=*/false, b,
      /*c_mod=*/(short)0, c, /*reuse_a=*/false, /*reuse_b=*/false);
}

// A fragment from row-major [16 x ldk] LDS tile, K = kbase..kbase+31.
// Order used here, same as the B pack (not the ISA table order): lane row = lane&15; per lane the 16 halves are one
// contiguous 16-half run at k = kbase+16*kg (+0..15).
__device__ __forceinline__ v16h load_a_frag(const _Float16* __restrict__ sA,
                                            int ldk, int kbase) {
  const int lane = threadIdx.x & 31;
  const int row  = lane & 15;
  const int kg   = (lane >> 4) & 1;
  const _Float16* p = sA + row * ldk + kbase + 16 * kg;   // 16B aligned
  v8h lo = *(const v8h*)p;          // ds_load_b128
  v8h hi = *(const v8h*)(p + 8);    // ds_load_b128
  return __builtin_shufflevector(lo, hi, 0, 1, 2, 3, 4, 5, 6, 7,
                                 8, 9, 10, 11, 12, 13, 14, 15);
}

// B fragment from PRE-PACKED weights: per (nt,kt) block, each lane's 16
// halves are stored contiguously -> one 32B per-lane load, wave-coalesced.
__device__ __forceinline__ v16h load_b_frag(const _Float16* __restrict__ P,
                                            int KT, int kt, int nt) {
  const int lane = threadIdx.x & 31;
  const _Float16* p = P + ((((size_t)nt * KT + kt) * 32 + lane) << 4);
  return *(const v16h*)p;           // 2x global_load_b128
}

// Pack row-major f32 weight [K x Nn] into f16 WMMA-B fragment layout.
// P[((nt*KT + kt)*32 + lane)*16 + e] = W[(kt*32 + 16*(lane>>4) + e)*Nn
//                                        + nt*16 + (lane&15)]
__global__ void pack_b_kernel(const float* __restrict__ W,
                              _Float16* __restrict__ P, int K, int Nn) {
  size_t i = (size_t)blockIdx.x * blockDim.x + threadIdx.x;
  if (i >= (size_t)K * Nn) return;
  int e    = (int)(i & 15);
  int lane = (int)((i >> 4) & 31);
  size_t blk = i >> 9;
  int KT = K >> 5;
  int kt = (int)(blk % KT);
  int nt = (int)(blk / KT);
  int col = nt * 16 + (lane & 15);
  int k   = kt * 32 + 16 * (lane >> 4) + e;
  P[i] = (_Float16)W[(size_t)k * Nn + col];
}

// ---------------- Edge message kernel: fused 3-stage MLP + scatter ----------
__global__ __launch_bounds__(128) void edge_msg_kernel(
    const float* __restrict__ hE, const float* __restrict__ hv,
    const int* __restrict__ Eidx, int Ecnt,
    const _Float16* __restrict__ W1, const float* __restrict__ B1,
    const _Float16* __restrict__ W2, const float* __restrict__ B2,
    const _Float16* __restrict__ W3, const float* __restrict__ B3,
    float* __restrict__ agg) {
  __shared__ __align__(16) _Float16 sA[16 * HD3];
  __shared__ __align__(16) _Float16 sM[16 * HD];
  __shared__ __align__(16) _Float16 sM2[16 * HD];
  const int e0 = blockIdx.x * 16;
  // Gather h_EV = [h_E | h_V[src] | h_V[dst]] tile -> LDS f16, float4 loads
  for (int idx = threadIdx.x; idx < 16 * (HD3 / 4); idx += 128) {
    int r = idx / (HD3 / 4);
    int c = (idx - r * (HD3 / 4)) * 4;
    int ge = e0 + r;
    float4 v = make_float4(0.f, 0.f, 0.f, 0.f);
    if (ge < Ecnt) {
      const float* sp;
      if (c < HD)          sp = hE + (size_t)ge * HD + c;
      else if (c < 2 * HD) sp = hv + (size_t)Eidx[ge] * HD + (c - HD);
      else                 sp = hv + (size_t)Eidx[Ecnt + ge] * HD + (c - 2 * HD);
      v = *(const float4*)sp;
    }
    v4h h;
    h[0] = (_Float16)v.x; h[1] = (_Float16)v.y;
    h[2] = (_Float16)v.z; h[3] = (_Float16)v.w;
    *(v4h*)(sA + (size_t)idx * 4) = h;   // ds_store_b64
  }
  __syncthreads();
  const int wave = threadIdx.x >> 5;
  const int lane = threadIdx.x & 31;
  const int c16  = lane & 15;
  const int kg   = (lane >> 4) & 1;
  const int nt0  = 2 * wave;  // 4 waves x 2 tiles = 8 N-tiles (128 cols)

  // Stage 1: [16,384] @ [384,128], relu
  v8f a0 = vzero(), a1 = vzero();
  for (int kt = 0; kt < HD3 / 32; ++kt) {
    v16h af = load_a_frag(sA, HD3, kt * 32);
    v16h bA = load_b_frag(W1, HD3 / 32, kt, nt0);
    v16h bB = load_b_frag(W1, HD3 / 32, kt, nt0 + 1);
    a0 = wmma16(af, bA, a0);
    a1 = wmma16(af, bB, a1);
  }
#pragma unroll
  for (int r = 0; r < 8; ++r) {
    int m = r + 8 * kg;
    int n0 = nt0 * 16 + c16, n1 = n0 + 16;
    sM[m * HD + n0] = (_Float16)fmaxf(a0[r] + B1[n0], 0.f);
    sM[m * HD + n1] = (_Float16)fmaxf(a1[r] + B1[n1], 0.f);
  }
  __syncthreads();

  // Stage 2: [16,128] @ [128,128], relu
  a0 = vzero(); a1 = vzero();
  for (int kt = 0; kt < HD / 32; ++kt) {
    v16h af = load_a_frag(sM, HD, kt * 32);
    v16h bA = load_b_frag(W2, HD / 32, kt, nt0);
    v16h bB = load_b_frag(W2, HD / 32, kt, nt0 + 1);
    a0 = wmma16(af, bA, a0);
    a1 = wmma16(af, bB, a1);
  }
#pragma unroll
  for (int r = 0; r < 8; ++r) {
    int m = r + 8 * kg;
    int n0 = nt0 * 16 + c16, n1 = n0 + 16;
    sM2[m * HD + n0] = (_Float16)fmaxf(a0[r] + B2[n0], 0.f);
    sM2[m * HD + n1] = (_Float16)fmaxf(a1[r] + B2[n1], 0.f);
  }
  __syncthreads();

  // Stage 3: [16,128] @ [128,128] + bias, scatter-sum to agg[src]
  a0 = vzero(); a1 = vzero();
  for (int kt = 0; kt < HD / 32; ++kt) {
    v16h af = load_a_frag(sM2, HD, kt * 32);
    v16h bA = load_b_frag(W3, HD / 32, kt, nt0);
    v16h bB = load_b_frag(W3, HD / 32, kt, nt0 + 1);
    a0 = wmma16(af, bA, a0);
    a1 = wmma16(af, bB, a1);
  }
#pragma unroll
  for (int r = 0; r < 8; ++r) {
    int m = r + 8 * kg;
    int ge = e0 + m;
    if (ge < Ecnt) {
      int sn = Eidx[ge];
      int n0 = nt0 * 16 + c16, n1 = n0 + 16;
      atomicAdd(&agg[(size_t)sn * HD + n0], a0[r] + B3[n0]);
      atomicAdd(&agg[(size_t)sn * HD + n1], a1[r] + B3[n1]);
    }
  }
}

// ---------------- Node update: agg/deg + LN1 + FFN + LN2 --------------------
__device__ __forceinline__ void block_ln(float* sX, _Float16* sXh,
                                         const float* __restrict__ g,
                                         const float* __restrict__ b,
                                         float* sRed, float* sStat) {
  const int tid = threadIdx.x;
  const int row = tid >> 4, c = tid & 15;  // 256 thr = 16 rows x 16 partials
  float s = 0.f;
  for (int j = 0; j < HD / 16; ++j) s += sX[row * HD + c * (HD / 16) + j];
  sRed[row * 16 + c] = s;
  __syncthreads();
  if (c == 0) {
    float t = 0.f;
    for (int j = 0; j < 16; ++j) t += sRed[row * 16 + j];
    sStat[row] = t * (1.f / HD);
  }
  __syncthreads();
  float mu = sStat[row];
  s = 0.f;
  for (int j = 0; j < HD / 16; ++j) {
    float d = sX[row * HD + c * (HD / 16) + j] - mu;
    s += d * d;
  }
  sRed[row * 16 + c] = s;
  __syncthreads();
  if (c == 0) {
    float t = 0.f;
    for (int j = 0; j < 16; ++j) t += sRed[row * 16 + j];
    sStat[16 + row] = rsqrtf(t * (1.f / HD) + 1e-5f);
  }
  __syncthreads();
  for (int idx = tid; idx < 16 * HD; idx += 256) {
    int r = idx / HD, cc = idx - r * HD;
    float v = (sX[idx] - sStat[r]) * sStat[16 + r] * g[cc] + b[cc];
    sX[idx] = v;
    if (sXh) sXh[idx] = (_Float16)v;
  }
  __syncthreads();
}

__global__ __launch_bounds__(256) void node_update_kernel(
    float* __restrict__ hv, const float* __restrict__ agg,
    const float* __restrict__ deg,
    const float* __restrict__ g1, const float* __restrict__ bb1,
    const _Float16* __restrict__ Wi, const float* __restrict__ Bi,
    const _Float16* __restrict__ Wo, const float* __restrict__ Bo,
    const float* __restrict__ g2, const float* __restrict__ bb2, int Ncnt) {
  __shared__ __align__(16) float    sX[16 * HD];
  __shared__ __align__(16) _Float16 sXh[16 * HD];
  __shared__ __align__(16) _Float16 sHid[16 * HD4];
  __shared__ float sRed[256];
  __shared__ float sStat[32];
  const int n0 = blockIdx.x * 16;
  for (int idx = threadIdx.x; idx < 16 * (HD / 4); idx += 256) {
    int r = idx / (HD / 4);
    int c = (idx - r * (HD / 4)) * 4;
    int gn = n0 + r;
    float4 v = make_float4(0.f, 0.f, 0.f, 0.f);
    if (gn < Ncnt) {
      float d = 1.f / fmaxf(deg[gn], 1.f);
      float4 a = *(const float4*)(hv + (size_t)gn * HD + c);
      float4 b = *(const float4*)(agg + (size_t)gn * HD + c);
      v = make_float4(a.x + b.x * d, a.y + b.y * d,
                      a.z + b.z * d, a.w + b.w * d);
    }
    *(float4*)(sX + (size_t)idx * 4) = v;
  }
  __syncthreads();
  block_ln(sX, sXh, g1, bb1, sRed, sStat);  // sX = LN1 out, sXh = f16 copy

  const int wave = threadIdx.x >> 5;
  const int lane = threadIdx.x & 31;
  const int c16  = lane & 15;
  const int kg   = (lane >> 4) & 1;

  // FFN GEMM1: [16,128]@[128,512]; 32 N-tiles, 8 waves -> 4 tiles each
  v8f acc[4];
#pragma unroll
  for (int t = 0; t < 4; ++t) acc[t] = vzero();
  for (int kt = 0; kt < HD / 32; ++kt) {
    v16h af = load_a_frag(sXh, HD, kt * 32);
#pragma unroll
    for (int t = 0; t < 4; ++t) {
      v16h bf = load_b_frag(Wi, HD / 32, kt, wave * 4 + t);
      acc[t] = wmma16(af, bf, acc[t]);
    }
  }
#pragma unroll
  for (int t = 0; t < 4; ++t) {
#pragma unroll
    for (int r = 0; r < 8; ++r) {
      int m = r + 8 * kg;
      int n = (wave * 4 + t) * 16 + c16;
      sHid[m * HD4 + n] = (_Float16)fmaxf(acc[t][r] + Bi[n], 0.f);
    }
  }
  __syncthreads();

  // FFN GEMM2: [16,512]@[512,128]; 8 N-tiles, 1 per wave
  v8f a2 = vzero();
  for (int kt = 0; kt < HD4 / 32; ++kt) {
    v16h af = load_a_frag(sHid, HD4, kt * 32);
    v16h bf = load_b_frag(Wo, HD4 / 32, kt, wave);
    a2 = wmma16(af, bf, a2);
  }
#pragma unroll
  for (int r = 0; r < 8; ++r) {
    int m = r + 8 * kg;
    int n = wave * 16 + c16;
    sX[m * HD + n] = sX[m * HD + n] + a2[r] + Bo[n];  // residual + dh
  }
  __syncthreads();
  block_ln(sX, (_Float16*)nullptr, g2, bb2, sRed, sStat);
  for (int idx = threadIdx.x; idx < 16 * (HD / 4); idx += 256) {
    int r = idx / (HD / 4);
    int c = (idx - r * (HD / 4)) * 4;
    int gn = n0 + r;
    if (gn < Ncnt)
      *(float4*)(hv + (size_t)gn * HD + c) = *(const float4*)(sX + (size_t)idx * 4);
  }
}

// ---------------- Projection head (B=16 => one WMMA M-tile) -----------------
__global__ __launch_bounds__(256) void proj_kernel(
    const float* __restrict__ pooled, const float* __restrict__ gcnt,
    const _Float16* __restrict__ P1, const _Float16* __restrict__ P2,
    const float* __restrict__ p2b, float* __restrict__ out, int Bcnt) {
  __shared__ __align__(16) _Float16 sXh[16 * HD];
  __shared__ __align__(16) _Float16 sHid[16 * HD];
  for (int idx = threadIdx.x; idx < 16 * HD; idx += 256) {
    int r = idx / HD, c = idx - r * HD;
    float v = 0.f;
    if (r < Bcnt) v = pooled[r * HD + c] / fmaxf(gcnt[r], 1.f);
    sXh[idx] = (_Float16)v;
  }
  __syncthreads();
  const int wave = threadIdx.x >> 5;
  const int lane = threadIdx.x & 31;
  const int c16  = lane & 15;
  const int kg   = (lane >> 4) & 1;
  v8f acc = vzero();
  for (int kt = 0; kt < HD / 32; ++kt) {
    v16h af = load_a_frag(sXh, HD, kt * 32);
    v16h bf = load_b_frag(P1, HD / 32, kt, wave);
    acc = wmma16(af, bf, acc);
  }
#pragma unroll
  for (int r = 0; r < 8; ++r) {
    int m = r + 8 * kg;
    int n = wave * 16 + c16;
    sHid[m * HD + n] = (_Float16)fmaxf(acc[r], 0.f);
  }
  __syncthreads();
  acc = vzero();
  for (int kt = 0; kt < HD / 32; ++kt) {
    v16h af = load_a_frag(sHid, HD, kt * 32);
    v16h bf = load_b_frag(P2, HD / 32, kt, wave);
    acc = wmma16(af, bf, acc);
  }
#pragma unroll
  for (int r = 0; r < 8; ++r) {
    int m = r + 8 * kg;
    int n = wave * 16 + c16;
    if (m < Bcnt) out[m * HD + n] = acc[r] + p2b[n];
  }
}

// ---------------- Small utility kernels -------------------------------------
__global__ void fill_zero_kernel(float* p, size_t n) {
  size_t i = (size_t)blockIdx.x * blockDim.x + threadIdx.x;
  if (i < n) p[i] = 0.f;
}
__global__ void copy_f32_kernel(const float* __restrict__ s, float* __restrict__ d, size_t n) {
  size_t i = (size_t)blockIdx.x * blockDim.x + threadIdx.x;
  if (i < n) d[i] = s[i];
}
__global__ void copy_i32_kernel(const int* __restrict__ s, int* __restrict__ d, size_t n) {
  size_t i = (size_t)blockIdx.x * blockDim.x + threadIdx.x;
  if (i < n) d[i] = s[i];
}
__global__ void deg_kernel(const int* __restrict__ Eidx, float* __restrict__ deg, int Ecnt) {
  int i = blockIdx.x * blockDim.x + threadIdx.x;
  if (i < Ecnt) atomicAdd(&deg[Eidx[i]], 1.f);
}
__global__ void gcnt_kernel(const int* __restrict__ bid, float* __restrict__ gcnt, int Ncnt) {
  int i = blockIdx.x * blockDim.x + threadIdx.x;
  if (i < Ncnt) atomicAdd(&gcnt[bid[i]], 1.f);
}
__global__ void pool_kernel(const float* __restrict__ hv, const int* __restrict__ bid,
                            float* __restrict__ pooled, int Ncnt) {
  size_t i = (size_t)blockIdx.x * blockDim.x + threadIdx.x;
  if (i < (size_t)Ncnt * HD) {
    int n = (int)(i / HD), c = (int)(i % HD);
    atomicAdd(&pooled[bid[n] * HD + c], hv[i]);
  }
}
__global__ void logits_kernel(const float* __restrict__ hv, const float* __restrict__ R,
                              const float* __restrict__ rb, float* __restrict__ out,
                              int Ncnt, int V) {
  size_t i = (size_t)blockIdx.x * blockDim.x + threadIdx.x;
  if (i < (size_t)Ncnt * V) {
    int n = (int)(i / V), v = (int)(i % V);
    float s = rb[v];
    const float* row = hv + (size_t)n * HD;
    for (int k = 0; k < HD; ++k) s += row[k] * R[k * V + v];
    out[i] = s;
  }
}

// ---------------- Host launch ------------------------------------------------
extern "C" void kernel_launch(void* const* d_in, const int* in_sizes, int n_in,
                              void* d_out, int out_size, void* d_ws, size_t ws_size,
                              hipStream_t stream) {
  (void)n_in; (void)ws_size;
  const float* h_V   = (const float*)d_in[0];
  const float* h_E   = (const float*)d_in[1];
  const int*   E_idx = (const int*)d_in[2];
  const int*   bid   = (const int*)d_in[3];
  const int*   S     = (const int*)d_in[4];
  const float* W1 = (const float*)d_in[5];  const float* b1 = (const float*)d_in[6];
  const float* W2 = (const float*)d_in[7];  const float* b2 = (const float*)d_in[8];
  const float* W3 = (const float*)d_in[9];  const float* b3 = (const float*)d_in[10];
  const float* ln1_g = (const float*)d_in[11]; const float* ln1_b = (const float*)d_in[12];
  const float* Wi = (const float*)d_in[13]; const float* bi = (const float*)d_in[14];
  const float* Wo = (const float*)d_in[15]; const float* bo = (const float*)d_in[16];
  const float* ln2_g = (const float*)d_in[17]; const float* ln2_b = (const float*)d_in[18];
  const float* P1 = (const float*)d_in[19]; const float* P2 = (const float*)d_in[20];
  const float* p2b = (const float*)d_in[21];
  const float* R  = (const float*)d_in[22]; const float* rb = (const float*)d_in[23];

  const int N = in_sizes[0] / HD;
  const int E = in_sizes[1] / HD;
  const int V = in_sizes[23];
  int Bq = (out_size - N * V - N) / HD;
  if (Bq < 1) Bq = 1;
  if (Bq > 16) Bq = 16;

  char* ws = (char*)d_ws;
  size_t off = 0;
  auto carve = [&](size_t bytes) -> char* {
    char* p = ws + off;
    off += (bytes + 255) & ~(size_t)255;
    return p;
  };
  float*     hv     = (float*)carve((size_t)N * HD * 4);
  float*     agg    = (float*)carve((size_t)N * HD * 4);
  float*     deg    = (float*)carve((size_t)N * 4);
  float*     pooled = (float*)carve((size_t)Bq * HD * 4);
  float*     gcnt   = (float*)carve((size_t)Bq * 4);
  _Float16*  w1h = (_Float16*)carve((size_t)LAY * HD3 * HD * 2);
  _Float16*  w2h = (_Float16*)carve((size_t)LAY * HD * HD * 2);
  _Float16*  w3h = (_Float16*)carve((size_t)LAY * HD * HD * 2);
  _Float16*  wih = (_Float16*)carve((size_t)LAY * HD * HD4 * 2);
  _Float16*  woh = (_Float16*)carve((size_t)LAY * HD4 * HD * 2);
  _Float16*  p1h = (_Float16*)carve((size_t)HD * HD * 2);
  _Float16*  p2h = (_Float16*)carve((size_t)HD * HD * 2);

  auto cdiv = [](size_t a, size_t b) -> unsigned { return (unsigned)((a + b - 1) / b); };

  // Pack weights into f16 WMMA-B fragment layout (once per launch; L2-resident)
  for (int l = 0; l < LAY; ++l) {
    pack_b_kernel<<<cdiv((size_t)HD3 * HD, 256), 256, 0, stream>>>(
        W1 + (size_t)l * HD3 * HD, w1h + (size_t)l * HD3 * HD, HD3, HD);
    pack_b_kernel<<<cdiv((size_t)HD * HD, 256), 256, 0, stream>>>(
        W2 + (size_t)l * HD * HD, w2h + (size_t)l * HD * HD, HD, HD);
    pack_b_kernel<<<cdiv((size_t)HD * HD, 256), 256, 0, stream>>>(
        W3 + (size_t)l * HD * HD, w3h + (size_t)l * HD * HD, HD, HD);
    pack_b_kernel<<<cdiv((size_t)HD * HD4, 256), 256, 0, stream>>>(
        Wi + (size_t)l * HD * HD4, wih + (size_t)l * HD * HD4, HD, HD4);
    pack_b_kernel<<<cdiv((size_t)HD4 * HD, 256), 256, 0, stream>>>(
        Wo + (size_t)l * HD4 * HD, woh + (size_t)l * HD4 * HD, HD4, HD);
  }
  pack_b_kernel<<<cdiv((size_t)HD * HD, 256), 256, 0, stream>>>(P1, p1h, HD, HD);
  pack_b_kernel<<<cdiv((size_t)HD * HD, 256), 256, 0, stream>>>(P2, p2h, HD, HD);

  copy_f32_kernel<<<cdiv((size_t)N * HD, 256), 256, 0, stream>>>(h_V, hv, (size_t)N * HD);
  fill_zero_kernel<<<cdiv((size_t)N, 256), 256, 0, stream>>>(deg, (size_t)N);
  deg_kernel<<<cdiv((size_t)E, 256), 256, 0, stream>>>(E_idx, deg, E);

  for (int l = 0; l < LAY; ++l) {
    fill_zero_kernel<<<cdiv((size_t)N * HD, 256), 256, 0, stream>>>(agg, (size_t)N * HD);
    edge_msg_kernel<<<cdiv((size_t)E, 16), 128, 0, stream>>>(
        h_E, hv, E_idx, E,
        w1h + (size_t)l * HD3 * HD, b1 + (size_t)l * HD,
        w2h + (size_t)l * HD * HD,  b2 + (size_t)l * HD,
        w3h + (size_t)l * HD * HD,  b3 + (size_t)l * HD, agg);
    node_update_kernel<<<cdiv((size_t)N, 16), 256, 0, stream>>>(
        hv, agg, deg, ln1_g + (size_t)l * HD, ln1_b + (size_t)l * HD,
        wih + (size_t)l * HD * HD4, bi + (size_t)l * HD4,
        woh + (size_t)l * HD4 * HD, bo + (size_t)l * HD,
        ln2_g + (size_t)l * HD, ln2_b + (size_t)l * HD, N);
  }

  fill_zero_kernel<<<cdiv((size_t)Bq * HD, 256), 256, 0, stream>>>(pooled, (size_t)Bq * HD);
  fill_zero_kernel<<<1, 256, 0, stream>>>(gcnt, (size_t)Bq);
  gcnt_kernel<<<cdiv((size_t)N, 256), 256, 0, stream>>>(bid, gcnt, N);
  pool_kernel<<<cdiv((size_t)N * HD, 256), 256, 0, stream>>>(hv, bid, pooled, N);

  float* out = (float*)d_out;
  logits_kernel<<<cdiv((size_t)N * V, 256), 256, 0, stream>>>(hv, R, rb, out, N, V);
  copy_i32_kernel<<<cdiv((size_t)N, 256), 256, 0, stream>>>(
      S, (int*)(out + (size_t)N * V), (size_t)N);
  proj_kernel<<<1, 256, 0, stream>>>(pooled, gcnt, p1h, p2h, p2b,
                                     out + (size_t)N * V + N, Bq);
}
